// Encoder_4088808866252
// MI455X (gfx1250) — matrix operand order repaired and checked by reading, not compiled, not run on hardware
//
#include <hip/hip_runtime.h>

typedef __attribute__((ext_vector_type(16))) _Float16 v16h;
typedef __attribute__((ext_vector_type(8)))  _Float16 v8h;
typedef __attribute__((ext_vector_type(8)))  float    v8f;
typedef __attribute__((ext_vector_type(4)))  unsigned int u32x4;
typedef __attribute__((ext_vector_type(8)))  int      i32x8;
typedef __attribute__((ext_vector_type(4)))  int      i32x4;

#define WMMA_F16(a,b,c) __builtin_amdgcn_wmma_f32_16x16x32_f16(false,(a),false,(b),(short)0,(c),false,false)

// B=2048, L=2, UNITS=16, DM=512, P=16 -> CH=1, D=32, C_ATT=32, C_FFN=128, HALF=8
#define EPS 1e-5f

// ---- TDM: DMA the 32KB per-sample activation (8192 f32) into LDS offset 0 --
// D# per CDNA5 ISA: group0 = {count|flags, lds_addr,
// global_addr lo, global_addr hi | type=2}; group1 packs data_size=4B,
// tensor_dim0 = tile_dim0 = 8192, tensor_dim1 = tile_dim1 = 1, stride = 8192.
__device__ inline void load_x_tdm(const float* gsrc, float* xs, int tid) {
#if __has_builtin(__builtin_amdgcn_tensor_load_to_lds)
  if (tid < 32) {                      // single wave issues the DMA
    unsigned long long ga = (unsigned long long)(const void*)gsrc;
    u32x4 g0;
    g0[0] = 1u;                                    // count=1, no gather/restore
    g0[1] = 0u;                                    // lds_addr = 0 (xs base)
    g0[2] = (unsigned)(ga & 0xffffffffu);          // global_addr[31:0]
    g0[3] = (unsigned)((ga >> 32) & 0x01ffffffu)   // global_addr[56:32]
          | (2u << 30);                            // type = 2 ("image")
    i32x8 g1 = {};
    g1[0] = (int)(2u << 16);                       // data_size = 2 (4 bytes)
    g1[1] = (int)(0x2000u << 16);                  // tensor_dim0 = 8192 (lo16)
    g1[2] = (int)(1u << 16);                       // dim0 hi=0 | tensor_dim1=1
    g1[3] = (int)(0x2000u << 16);                  // dim1 hi=0 | tile_dim0=8192
    g1[4] = 1;                                     // tile_dim1 = 1
    g1[5] = 0x2000;                                // tensor_dim0_stride = 8192
    i32x4 z4 = {};
#if __clang_major__ >= 23
    i32x8 z8 = {};
    __builtin_amdgcn_tensor_load_to_lds(g0, g1, z4, z4, z8, 0);
#else
    __builtin_amdgcn_tensor_load_to_lds(g0, g1, z4, z4, 0);
#endif
    __builtin_amdgcn_s_wait_tensorcnt(0);
  }
  (void)xs;
  __syncthreads();
#else
  for (int i = tid; i < 8192; i += 256) xs[i] = gsrc[i];
  __syncthreads();
#endif
}

// ---- WMMA fragment loaders (contiguous k = 16h + s order, wave32) -----------------------
// A (16x32 f16): lane&15 = row; lanes 0-15 hold K k0..k0+15,
// lanes 16-31 hold K k0+16..k0+31. Two aligned 16B LDS loads.
__device__ inline v16h ldA(const _Float16* src, int ld, int m0, int k0, int lane) {
  int r = m0 + (lane & 15); int kb = k0 + ((lane & 16) ? 16 : 0);
  const v8h* p = (const v8h*)(src + r * ld + kb);   // ld mult of 8, kb mult of 16
  v8h lo = p[0], hi = p[1];                          // halves [0..7], [8..15]
  v16h a;
#pragma unroll
  for (int j = 0; j < 8; ++j) { a[j] = lo[j]; a[j + 8] = hi[j]; }
  return a;
}
// B (32x16 f16) from a TRANSPOSED store (src[col][k], K contiguous):
// lane&15 = column; lanes 0-15 take K rows k0..k0+15, lanes 16-31 k0+16..k0+31.
__device__ inline v16h ldBT(const _Float16* src, int ld, int k0, int n0, int lane) {
  int c = n0 + (lane & 15); int kb = k0 + ((lane & 16) ? 16 : 0);
  const v8h* p = (const v8h*)(src + c * ld + kb);   // ld mult of 8, kb mult of 16
  v8h lo = p[0], hi = p[1];
  v16h b;
#pragma unroll
  for (int j = 0; j < 8; ++j) { b[j] = lo[j]; b[j + 8] = hi[j]; }
  return b;
}
// B from transposed f32 global weights (lin_w[o][d] used as B[d][o]).
// 32-bit offset off a uniform base -> SADDR+voffset addressing.
__device__ inline v16h ldBTf32(const float* src, int ld, int k0, int n0, int lane) {
  int c = n0 + (lane & 15); int kb = k0 + ((lane & 16) ? 16 : 0);
  int off = c * ld + kb;
  v16h b;
#pragma unroll
  for (int j = 0; j < 16; ++j) b[j] = (_Float16)src[off + j];
  return b;
}

__device__ inline float wred_sum(float v) {
  for (int o = 16; o; o >>= 1) v += __shfl_xor(v, o, 32);
  return v;
}
__device__ inline float wred_max(float v) {
  for (int o = 16; o; o >>= 1) v = fmaxf(v, __shfl_xor(v, o, 32));
  return v;
}

// ===========================================================================
// Attention sublayer: x += attn(LN(x)).  One workgroup (8 wave32) per sample.
// LDS (84 KB): xs f32[8192] | Dbuf f32[2048] | hp h[8192] | cvT h[2048]
//              | pw16 h[1024] | q16/k16 h[2048]x2 | vT h[2048] | sc16 h[1024]
//              | t16 h[4096]
// ===========================================================================
__global__ __launch_bounds__(256) void attn_kernel(
    float* __restrict__ x,
    const float* __restrict__ ln1g, const float* __restrict__ ln1b,
    const float* __restrict__ qkv_dw,
    const float* __restrict__ bng, const float* __restrict__ bnb,
    const float* __restrict__ bnm, const float* __restrict__ bnv,
    const float* __restrict__ qkv_pw,
    const float* __restrict__ position,
    const float* __restrict__ c1dw, const float* __restrict__ c1db,
    const float* __restrict__ linw, const float* __restrict__ linb,
    int l) {
  extern __shared__ char smem[];
  float*    xs   = (float*)smem;                  // 32768 B (must stay at 0: TDM)
  float*    Dbuf = (float*)(smem + 32768);        //  8192 B
  _Float16* hp   = (_Float16*)(smem + 40960);     // 16384 B  patches [c=32][s=256]
  _Float16* cvT  = (_Float16*)(smem + 57344);     //  4096 B  conv out  [s=64][c=32]
  _Float16* pw16 = (_Float16*)(smem + 61440);     //  2048 B  pw weights [32][32]
  _Float16* q16  = (_Float16*)(smem + 63488);     //  4096 B  [d=32][e=64]
  _Float16* k16  = (_Float16*)(smem + 67584);     //  4096 B  [f=32][e=64]
  _Float16* vT   = (_Float16*)(smem + 71680);     //  4096 B  [e=64][f=32]
  _Float16* sc16 = (_Float16*)(smem + 75776);     //  2048 B  softmax [32][32]
  _Float16* t16  = (_Float16*)(smem + 77824);     //  8192 B  c1d out [16][256]

  const int tid = threadIdx.x, lane = tid & 31, wave = tid >> 5;
  const int b = blockIdx.x;
  const float* gx = x + (size_t)b * 8192;

  load_x_tdm(gx, xs, tid);

  // ---- LayerNorm + to_patches (x[u][m] -> hp[d=m/16][h=u][w=m%16], f16) ----
  const float* g  = ln1g + l * 512;
  const float* be = ln1b + l * 512;
  for (int rr = 0; rr < 2; ++rr) {
    int r = wave * 2 + rr;
    float s = 0.f, s2 = 0.f;
    for (int i = lane; i < 512; i += 32) { float v = xs[r * 512 + i]; s += v; s2 += v * v; }
    s = wred_sum(s); s2 = wred_sum(s2);
    float mean = s * (1.f / 512.f);
    float rstd = rsqrtf(s2 * (1.f / 512.f) - mean * mean + EPS);
    for (int i = lane; i < 512; i += 32) {
      float v = (xs[r * 512 + i] - mean) * rstd * g[i] + be[i];
      hp[(i >> 4) * 256 + r * 16 + (i & 15)] = (_Float16)v;
    }
  }
  __syncthreads();

  // ---- q/k/v: dw3x3 stride2 + BN (VALU) then 32x32 pointwise (WMMA) -------
#pragma unroll
  for (int i = 0; i < 3; ++i) {   // i is compile-time: keeps LDS addrspace ptrs
    const float* dw = qkv_dw + (size_t)((l * 3 + i) * 32) * 9;
    const float* bg = bng + (l * 3 + i) * 32; const float* bb = bnb + (l * 3 + i) * 32;
    const float* bm = bnm + (l * 3 + i) * 32; const float* bv = bnv + (l * 3 + i) * 32;
    for (int e = tid; e < 2048; e += 256) {
      int c = e >> 6, s = e & 63, oh = s >> 3, ow = s & 7;
      float acc = 0.f;
      const float* wp = dw + c * 9;
      for (int dy = 0; dy < 3; ++dy) {
        int yy = oh * 2 + dy - 1; if ((unsigned)yy >= 16u) continue;
        for (int dx = 0; dx < 3; ++dx) {
          int xx = ow * 2 + dx - 1; if ((unsigned)xx >= 16u) continue;
          acc += (float)hp[c * 256 + yy * 16 + xx] * wp[dy * 3 + dx];
        }
      }
      float sc = bg[c] * rsqrtf(bv[c] + EPS);
      cvT[s * 32 + c] = (_Float16)((acc - bm[c]) * sc + bb[c]);   // transposed
    }
    const float* pw = qkv_pw + (size_t)(l * 3 + i) * 1024;
    for (int e = tid; e < 1024; e += 256) pw16[e] = (_Float16)pw[e];
    __syncthreads();
    // OUT[32x64] = pw[32x32] @ cv[32x64]   (2x4 tiles, K=32)
    for (int t = wave; t < 8; t += 8) {
      int m0 = (t >> 2) << 4, n0 = (t & 3) << 4;
      v8f acc = {};
      v16h a  = ldA(pw16, 32, m0, 0, lane);
      v16h bf = ldBT(cvT, 32, 0, n0, lane);
      acc = WMMA_F16(a, bf, acc);
      int col = n0 + (lane & 15), rb = m0 + ((lane & 16) ? 8 : 0);
      if (i == 2) {                       // v stored [e][f]: packed 16B store
        v8h hv;
#pragma unroll
        for (int j = 0; j < 8; ++j) hv[j] = (_Float16)acc[j];
        *(v8h*)(vT + col * 32 + rb) = hv;
      } else {                            // q,k stored [ch][e]: strided rows
#pragma unroll
        for (int j = 0; j < 8; ++j)
          (q16 + i * 2048)[(rb + j) * 64 + col] = (_Float16)acc[j];
      }
    }
    __syncthreads();
  }

  // ---- scores[32x32] = q @ k^T / 8 + position (WMMA, 2x2 tiles, K=64) -----
  const float* pos = position + (size_t)l * 1024;
  for (int t = wave; t < 4; t += 8) {
    int m0 = (t >> 1) << 4, n0 = (t & 1) << 4;
    v8f acc = {};
    for (int kk = 0; kk < 2; ++kk) {
      v16h a  = ldA(q16, 64, m0, kk * 32, lane);
      v16h bf = ldBT(k16, 64, kk * 32, n0, lane);   // k16[f][e]: K=e contiguous
      acc = WMMA_F16(a, bf, acc);
    }
    int col = n0 + (lane & 15), rb = m0 + ((lane & 16) ? 8 : 0);
#pragma unroll
    for (int j = 0; j < 8; ++j) {
      int r = rb + j;
      Dbuf[r * 32 + col] = acc[j] * 0.125f + pos[r * 32 + col];
    }
  }
  __syncthreads();

  // ---- softmax over rows (lane-parallel) ----------------------------------
  for (int r = wave; r < 32; r += 8) {
    float v = Dbuf[r * 32 + lane];
    float m = wred_max(v);
    float e = expf(v - m);
    float s = wred_sum(e);
    sc16[r * 32 + lane] = (_Float16)(e / s);
  }
  __syncthreads();

  // ---- o[32x64] = attn @ v (WMMA, 2x4 tiles, K=32) ------------------------
  for (int t = wave; t < 8; t += 8) {
    int m0 = (t >> 2) << 4, n0 = (t & 3) << 4;
    v8f acc = {};
    v16h a  = ldA(sc16, 32, m0, 0, lane);
    v16h bf = ldBT(vT, 32, 0, n0, lane);            // vT[e][f]: K=f contiguous
    acc = WMMA_F16(a, bf, acc);
    int col = n0 + (lane & 15), rb = m0 + ((lane & 16) ? 8 : 0);
#pragma unroll
    for (int j = 0; j < 8; ++j) Dbuf[(rb + j) * 64 + col] = acc[j];
  }
  __syncthreads();

  // ---- reshape + c1d (16x8 weights, VALU): t16[u][d*8+p2] -----------------
  const float* cw = c1dw + l * 16 * 8;
  const float* cb = c1db + l * 16;
  for (int e = tid; e < 4096; e += 256) {
    int u = e >> 8, j = e & 255, d = j >> 3, p2 = j & 7;
    float acc = cb[u];
    for (int c = 0; c < 8; ++c) acc += Dbuf[d * 64 + c * 8 + p2] * cw[u * 8 + c];
    t16[u * 256 + j] = (_Float16)acc;
  }
  __syncthreads();

  // ---- lin: xs[16x512] += t16[16x256] @ lin_w^T + lin_b (WMMA) ------------
  const float* lw = linw + (size_t)l * 512 * 256;
  const float* lb = linb + l * 512;
  for (int t = wave; t < 32; t += 8) {
    int n0 = t << 4;
    v8f acc = {};
    for (int kk = 0; kk < 8; ++kk) {
      v16h a  = ldA(t16, 256, 0, kk * 32, lane);
      v16h bf = ldBTf32(lw, 256, kk * 32, n0, lane);
      acc = WMMA_F16(a, bf, acc);
    }
    int col = n0 + (lane & 15), rb = ((lane & 16) ? 8 : 0);
#pragma unroll
    for (int j = 0; j < 8; ++j) xs[(rb + j) * 512 + col] += acc[j] + lb[col];
  }
  __syncthreads();

  float* ox = x + (size_t)b * 8192;
  for (int i = tid; i < 8192; i += 256) ox[i] = xs[i];
}

// ===========================================================================
// Conv-FFN sublayer: x += from_patches(ffn(LN(x))).  208 KB LDS (CDNA5 WGP
// has 320 KB, so the full 128x256 f16 intermediates stay resident).
// LDS: xs f32[8192] | hpT h[8192] (patches [s][c], later f2 weights) |
//      y1 h[32768] (f1 out [c][s], later y3T [s][c]) | y2T h[32768] [s][c] |
//      wb h[16384]
// ===========================================================================
__global__ __launch_bounds__(256) void ffn_kernel(
    float* __restrict__ x,
    const float* __restrict__ ln2g, const float* __restrict__ ln2b,
    const float* __restrict__ f1w, const float* __restrict__ f1b,
    const float* __restrict__ fdww, const float* __restrict__ fdwb,
    const float* __restrict__ bng, const float* __restrict__ bnb,
    const float* __restrict__ bnm, const float* __restrict__ bnv,
    const float* __restrict__ fpww, const float* __restrict__ fpwb,
    const float* __restrict__ f2w, const float* __restrict__ f2b,
    int l) {
  extern __shared__ char smem[];
  float*    xs  = (float*)smem;                   // 32768 B (must stay at 0: TDM)
  _Float16* hpT = (_Float16*)(smem + 32768);      // 16384 B  [s=256][c=32]
  _Float16* y1  = (_Float16*)(smem + 49152);      // 65536 B  [c=128][s=256]
  _Float16* y2T = (_Float16*)(smem + 114688);     // 65536 B  [s=256][c=128]
  _Float16* wb  = (_Float16*)(smem + 180224);     // 32768 B
  _Float16* y3T = y1;                              // reuse after dw conv
  _Float16* f2w16 = hpT;                           // reuse after f1

  const int tid = threadIdx.x, lane = tid & 31, wave = tid >> 5;
  const int b = blockIdx.x;
  const float* gx = x + (size_t)b * 8192;

  load_x_tdm(gx, xs, tid);

  // ---- LayerNorm + to_patches (transposed [s][c]); stage f1 weights -------
  const float* g  = ln2g + l * 512;
  const float* be = ln2b + l * 512;
  for (int rr = 0; rr < 2; ++rr) {
    int r = wave * 2 + rr;
    float s = 0.f, s2 = 0.f;
    for (int i = lane; i < 512; i += 32) { float v = xs[r * 512 + i]; s += v; s2 += v * v; }
    s = wred_sum(s); s2 = wred_sum(s2);
    float mean = s * (1.f / 512.f);
    float rstd = rsqrtf(s2 * (1.f / 512.f) - mean * mean + EPS);
    for (int i = lane; i < 512; i += 32) {
      float v = (xs[r * 512 + i] - mean) * rstd * g[i] + be[i];
      hpT[(r * 16 + (i & 15)) * 32 + (i >> 4)] = (_Float16)v;
    }
  }
  const float* w1 = f1w + (size_t)l * 128 * 32;
  for (int e = tid; e < 4096; e += 256) wb[e] = (_Float16)w1[e];
  __syncthreads();

  // ---- f1: y1[128x256] = relu(W1[128x32] @ patches[32x256] + b1) (WMMA) ---
  const float* fb1 = f1b + l * 128;
  for (int t = wave; t < 128; t += 8) {
    int m0 = (t >> 4) << 4, n0 = (t & 15) << 4;
    v8f acc = {};
    v16h a  = ldA(wb, 32, m0, 0, lane);
    v16h bf = ldBT(hpT, 32, 0, n0, lane);
    acc = WMMA_F16(a, bf, acc);
    int col = n0 + (lane & 15), rb = m0 + ((lane & 16) ? 8 : 0);
#pragma unroll
    for (int j = 0; j < 8; ++j) {
      int r = rb + j;
      y1[r * 256 + col] = (_Float16)fmaxf(acc[j] + fb1[r], 0.f);
    }
  }
  __syncthreads();

  // ---- depthwise 3x3 + bias + BN -> y2T (VALU); stage fpw/f2 weights ------
  const float* dwp = fdww + (size_t)l * 128 * 9;
  const float* db  = fdwb + l * 128;
  const float* bg = bng + l * 128; const float* bb = bnb + l * 128;
  const float* bm = bnm + l * 128; const float* bv = bnv + l * 128;
  for (int e = tid; e < 32768; e += 256) {
    int c = e >> 8, s = e & 255, h = s >> 4, w = s & 15;
    float acc = db[c];
    const float* wp = dwp + c * 9;
    for (int dy = 0; dy < 3; ++dy) {
      int yy = h + dy - 1; if ((unsigned)yy >= 16u) continue;
      for (int dx = 0; dx < 3; ++dx) {
        int xx = w + dx - 1; if ((unsigned)xx >= 16u) continue;
        acc += (float)y1[c * 256 + yy * 16 + xx] * wp[dy * 3 + dx];
      }
    }
    float sc = bg[c] * rsqrtf(bv[c] + EPS);
    y2T[s * 128 + c] = (_Float16)((acc - bm[c]) * sc + bb[c]);   // transposed
  }
  const float* pw = fpww + (size_t)l * 128 * 128;
  for (int e = tid; e < 16384; e += 256) wb[e] = (_Float16)pw[e];     // fpw weights
  const float* w2 = f2w + (size_t)l * 32 * 128;
  for (int e = tid; e < 4096; e += 256) f2w16[e] = (_Float16)w2[e];   // hp region dead
  __syncthreads();

  // ---- fpw: y3T = relu(Wpw[128x128] @ y2 + bpw) (WMMA, into y1 region) ----
  const float* pb = fpwb + l * 128;
  for (int t = wave; t < 128; t += 8) {
    int m0 = (t >> 4) << 4, n0 = (t & 15) << 4;
    v8f acc = {};
    for (int kk = 0; kk < 4; ++kk) {
      v16h a  = ldA(wb, 128, m0, kk * 32, lane);
      v16h bf = ldBT(y2T, 128, kk * 32, n0, lane);
      acc = WMMA_F16(a, bf, acc);
    }
    int col = n0 + (lane & 15), rb = m0 + ((lane & 16) ? 8 : 0);
    v8h hv;                              // [s][c]: 8 adjacent channels -> 16B
#pragma unroll
    for (int j = 0; j < 8; ++j) hv[j] = (_Float16)fmaxf(acc[j] + pb[rb + j], 0.f);
    *(v8h*)(y3T + col * 128 + rb) = hv;
  }
  __syncthreads();

  // ---- f2 + from_patches residual: xs[h][d*16+w] += (W2 @ y3 + b2) (WMMA) -
  const float* b2 = f2b + l * 32;
  for (int t = wave; t < 32; t += 8) {
    int m0 = (t >> 4) << 4, n0 = (t & 15) << 4;
    v8f acc = {};
    for (int kk = 0; kk < 4; ++kk) {
      v16h a  = ldA(f2w16, 128, m0, kk * 32, lane);
      v16h bf = ldBT(y3T, 128, kk * 32, n0, lane);
      acc = WMMA_F16(a, bf, acc);
    }
    int col = n0 + (lane & 15), rb = m0 + ((lane & 16) ? 8 : 0);
    int h = col >> 4, w = col & 15;
#pragma unroll
    for (int j = 0; j < 8; ++j) {
      int d = rb + j;
      xs[h * 512 + d * 16 + w] += acc[j] + b2[d];
    }
  }
  __syncthreads();

  float* ox = x + (size_t)b * 8192;
  for (int i = tid; i < 8192; i += 256) ox[i] = xs[i];
}

// ===========================================================================
extern "C" void kernel_launch(void* const* d_in, const int* in_sizes, int n_in,
                              void* d_out, int out_size, void* d_ws, size_t ws_size,
                              hipStream_t stream) {
  (void)in_sizes; (void)n_in; (void)out_size; (void)d_ws; (void)ws_size;
  const float* in_x = (const float*)d_in[0];
  float* x = (float*)d_out;   // residual stream lives in d_out
  hipMemcpyAsync(x, in_x, (size_t)2048 * 16 * 512 * sizeof(float),
                 hipMemcpyDeviceToDevice, stream);

  const float* ln1g = (const float*)d_in[1];  const float* ln1b = (const float*)d_in[2];
  const float* ln2g = (const float*)d_in[3];  const float* ln2b = (const float*)d_in[4];
  const float* qdw  = (const float*)d_in[5];
  const float* qbng = (const float*)d_in[6];  const float* qbnb = (const float*)d_in[7];
  const float* qbnm = (const float*)d_in[8];  const float* qbnv = (const float*)d_in[9];
  const float* qpw  = (const float*)d_in[10];
  const float* pos  = (const float*)d_in[11];
  const float* c1dw = (const float*)d_in[12]; const float* c1db = (const float*)d_in[13];
  const float* linw = (const float*)d_in[14]; const float* linb = (const float*)d_in[15];
  const float* f1w  = (const float*)d_in[16]; const float* f1b  = (const float*)d_in[17];
  const float* fdww = (const float*)d_in[18]; const float* fdwb = (const float*)d_in[19];
  const float* fbng = (const float*)d_in[20]; const float* fbnb = (const float*)d_in[21];
  const float* fbnm = (const float*)d_in[22]; const float* fbnv = (const float*)d_in[23];
  const float* fpww = (const float*)d_in[24]; const float* fpwb = (const float*)d_in[25];
  const float* f2w  = (const float*)d_in[26]; const float* f2b  = (const float*)d_in[27];

  const size_t SM_ATTN = 86016;    //  84 KB dynamic LDS
  const size_t SM_FFN  = 212992;   // 208 KB dynamic LDS (CDNA5 WGP: 320 KB)
  for (int l = 0; l < 2; ++l) {
    attn_kernel<<<2048, 256, SM_ATTN, stream>>>(
        x, ln1g, ln1b, qdw, qbng, qbnb, qbnm, qbnv, qpw, pos,
        c1dw, c1db, linw, linb, l);
    ffn_kernel<<<2048, 256, SM_FFN, stream>>>(
        x, ln2g, ln2b, f1w, f1b, fdww, fdwb, fbng, fbnb, fbnm, fbnv,
        fpww, fpwb, f2w, f2b, l);
  }
}
